// MultiHeadedAttention_6416681140387
// MI455X (gfx1250) — matrix operand order repaired and checked by reading, not compiled, not run on hardware
//
#include <hip/hip_runtime.h>
#include <hip/hip_bf16.h>

// ---------------- problem constants (shapes fixed by setup_inputs) ----------
#define BT    8
#define T4    4
#define C256  256
#define H96   96
#define W96   96
#define HW    9216         // 96*96
#define N0    2304         // t*24*24
#define D0    2048         // 128*16
#define N1    576          // t*12*12
#define D1    8192         // 128*64
#define CONVK 2304         // 256*9

// ---- WMMA tile core: BM=BN=128, BK=32, 256 thr = 8 waves (4Mx2N),
// ---- each wave owns a 32x64 output tile (2x4 WMMA tiles), double-buffered LDS.
#define LDA 40             // halves per A-row in LDS (32 + 8 pad), 80B, 16B-aligned rows
#define LDB 136            // halves per B-row in LDS (128 + 8 pad), 272B, 16B-aligned rows
#define A_TILE (128 * LDA) // 5120 halves
#define B_TILE (32 * LDB)  // 4352 halves

typedef __attribute__((ext_vector_type(16))) _Float16 v16h;
typedef __attribute__((ext_vector_type(8)))  _Float16 v8h;
typedef __attribute__((ext_vector_type(8)))  float    v8f;

#define WMMA_F16(acc, a, b) \
    acc = __builtin_amdgcn_wmma_f32_16x16x32_f16(false, a, false, b, (short)0, acc, false, false)

// A fragment 16x32 f16: lane L -> row M = L&15, half h = L>>4.
// K runs: [8h, 8h+8) and [16+8h, 16+8h+8)  => two ds_load_b128.
__device__ __forceinline__ v16h load_frag_a(const _Float16* As, int mOff, int lane) {
    const _Float16* p = As + (mOff + (lane & 15)) * LDA + ((lane >> 4) << 3);
    v8h lo = *(const v8h*)(p);
    v8h hi = *(const v8h*)(p + 16);
    v16h r;
#pragma unroll
    for (int i = 0; i < 8; ++i) { r[i] = lo[i]; r[i + 8] = hi[i]; }
    return r;
}

// B fragment 32x16 f16: lane L -> col N = L&15, half h = L>>4; K runs [8h, 8h+8) and [16+8h, 16+8h+8).
__device__ __forceinline__ v16h load_frag_b(const _Float16* Bs, int nOff, int lane) {
    const _Float16* p = Bs + ((lane >> 4) << 3) * LDB + nOff + (lane & 15);
    v16h r;
#pragma unroll
    for (int i = 0; i < 8; ++i) { r[i] = p[i * LDB]; r[i + 8] = p[(16 + i) * LDB]; }
    return r;
}

// one K-step of the 128x128 block tile: 2 A frags x 4 B frags = 8 WMMAs
__device__ __forceinline__ void mma_step(const _Float16* As, const _Float16* Bs,
                                         int mOff, int nOff, int lane, v8f (&acc)[2][4]) {
    v16h a0 = load_frag_a(As, mOff, lane);
    v16h a1 = load_frag_a(As, mOff + 16, lane);
    v16h b0 = load_frag_b(Bs, nOff, lane);
    v16h b1 = load_frag_b(Bs, nOff + 16, lane);
    v16h b2 = load_frag_b(Bs, nOff + 32, lane);
    v16h b3 = load_frag_b(Bs, nOff + 48, lane);
    WMMA_F16(acc[0][0], a0, b0); WMMA_F16(acc[0][1], a0, b1);
    WMMA_F16(acc[0][2], a0, b2); WMMA_F16(acc[0][3], a0, b3);
    WMMA_F16(acc[1][0], a1, b0); WMMA_F16(acc[1][1], a1, b1);
    WMMA_F16(acc[1][2], a1, b2); WMMA_F16(acc[1][3], a1, b3);
}

__device__ __forceinline__ void st4h(_Float16* p, float4 f) {
    p[0] = (_Float16)f.x; p[1] = (_Float16)f.y;
    p[2] = (_Float16)f.z; p[3] = (_Float16)f.w;
}

// window <-> tensor index: row n -> (tt,oh,ow), feature d -> (c,hh,ww)
__device__ __forceinline__ int win_addr(int n, int d, int img0, int chOff,
                                        int perT, int outW, int winW, int winH,
                                        int aSh, int wSh) {
    int tt  = n / perT;
    int rem = n - tt * perT;
    int oh  = rem / outW;
    int ow  = rem - oh * outW;
    int cc  = d >> aSh;
    int hh  = (d & ((1 << aSh) - 1)) >> wSh;
    int wwi = d & (winW - 1);
    int h = oh * winH + hh;
    int w = ow * winW + wwi;
    return ((img0 + tt) * C256 + chOff + cc) * HW + h * W96 + w;
}

// ============ K1: fused QKV 1x1-conv GEMM (M=256,N=9216,K=256), f16 out =====
__global__ __launch_bounds__(256)
void qkv_gemm_kernel(const float* __restrict__ x,
                     const float* __restrict__ wq, const float* __restrict__ bq,
                     const float* __restrict__ wk, const float* __restrict__ bk,
                     const float* __restrict__ wv, const float* __restrict__ bv,
                     _Float16* __restrict__ qf, _Float16* __restrict__ kf,
                     _Float16* __restrict__ vf) {
    __shared__ _Float16 As[2 * A_TILE];
    __shared__ _Float16 Bs[2 * B_TILE];

    const int tid  = threadIdx.x;
    const int lane = tid & 31;
    const int wave = tid >> 5;
    const int mOff = (wave >> 1) << 5;   // 0,32,64,96
    const int nOff = (wave & 1) << 6;    // 0,64

    const int img   = blockIdx.z & 7;
    const int which = blockIdx.z >> 3;   // 0=q 1=k 2=v
    const int mBase = blockIdx.y << 7;
    const int nBase = blockIdx.x << 7;

    const float* Wm   = (which == 0) ? wq : (which == 1) ? wk : wv;
    const float* bias = (which == 0) ? bq : (which == 1) ? bk : bv;
    _Float16*    outp = ((which == 0) ? qf : (which == 1) ? kf : vf) + img * C256 * HW;
    const float* X    = x + img * C256 * HW;

    auto stage = [&](int k0, int buf) {
        _Float16* Ad = As + buf * A_TILE;
        _Float16* Bd = Bs + buf * B_TILE;
#pragma unroll
        for (int i = 0; i < 4; ++i) {             // A: 128x32 fp32 -> f16
            int idx = tid + (i << 8);
            int row = idx >> 3, c4 = (idx & 7) << 2;
            float4 f = *(const float4*)(Wm + (mBase + row) * 256 + k0 + c4);
            st4h(&Ad[row * LDA + c4], f);
        }
#pragma unroll
        for (int i = 0; i < 4; ++i) {             // B: 32x128 fp32 -> f16
            int idx = tid + (i << 8);
            int row = idx >> 5, c4 = (idx & 31) << 2;
            float4 f = *(const float4*)(X + (k0 + row) * HW + nBase + c4);
            st4h(&Bd[row * LDB + c4], f);
        }
    };

    v8f acc[2][4] = {};
    stage(0, 0);
    __syncthreads();
    const int nT = 256 / 32;
    for (int t = 0; t < nT; ++t) {
        if (t + 1 < nT) {
            __builtin_prefetch(X + (32 * (t + 1)) * HW + nBase, 0, 1);
            stage(32 * (t + 1), (t + 1) & 1);
        }
        mma_step(As + (t & 1) * A_TILE, Bs + (t & 1) * B_TILE, mOff, nOff, lane, acc);
        __syncthreads();
    }

    const int hi = (lane >> 4) << 3;
    const int nc = lane & 15;
#pragma unroll
    for (int mi = 0; mi < 2; ++mi)
#pragma unroll
        for (int r = 0; r < 8; ++r) {
            int gm = mBase + mOff + (mi << 4) + r + hi;
            float bb = bias[gm];
#pragma unroll
            for (int ni = 0; ni < 4; ++ni)
                outp[gm * HW + nBase + nOff + (ni << 4) + nc] = (_Float16)(acc[mi][ni][r] + bb);
        }
}

// ============ K2: windowed attention scores  S = Q Kt * scale (fp32 out) ====
__global__ __launch_bounds__(256)
void attn_scores_kernel(const _Float16* __restrict__ qf, const _Float16* __restrict__ kf,
                        float* __restrict__ S, int N, int D, int chOff, int perT,
                        int outW, int winW, int winH, int aSh, int wSh, float scale) {
    __shared__ _Float16 As[2 * A_TILE];
    __shared__ _Float16 Bs[2 * B_TILE];

    const int tid  = threadIdx.x;
    const int lane = tid & 31;
    const int wave = tid >> 5;
    const int mOff = (wave >> 1) << 5;
    const int nOff = (wave & 1) << 6;

    const int bIdx  = blockIdx.z;
    const int img0  = bIdx * T4;
    const int mBase = blockIdx.y << 7;
    const int nBase = blockIdx.x << 7;

    auto stage = [&](int k0, int buf) {
        _Float16* Ad = As + buf * A_TILE;
        _Float16* Bd = Bs + buf * B_TILE;
#pragma unroll
        for (int i = 0; i < 16; ++i) {            // A: Q rows (gather, clamp rows)
            int idx = tid + (i << 8);
            int row = idx >> 5, dd = idx & 31;
            int gr = mBase + row; if (gr >= N) gr = N - 1;
            Ad[row * LDA + dd] =
                qf[win_addr(gr, k0 + dd, img0, chOff, perT, outW, winW, winH, aSh, wSh)];
        }
#pragma unroll
        for (int i = 0; i < 16; ++i) {            // B: K rows transposed (gather, clamp cols)
            int idx = tid + (i << 8);
            int kk = idx >> 7, nn = idx & 127;
            int gc = nBase + nn; if (gc >= N) gc = N - 1;
            Bd[kk * LDB + nn] =
                kf[win_addr(gc, k0 + kk, img0, chOff, perT, outW, winW, winH, aSh, wSh)];
        }
    };

    v8f acc[2][4] = {};
    stage(0, 0);
    __syncthreads();
    const int nT = D / 32;
    for (int t = 0; t < nT; ++t) {
        if (t + 1 < nT) stage(32 * (t + 1), (t + 1) & 1);
        mma_step(As + (t & 1) * A_TILE, Bs + (t & 1) * B_TILE, mOff, nOff, lane, acc);
        __syncthreads();
    }

    float* So = S + bIdx * N * N;
    const int hi = (lane >> 4) << 3;
    const int nc = lane & 15;
#pragma unroll
    for (int mi = 0; mi < 2; ++mi)
#pragma unroll
        for (int r = 0; r < 8; ++r) {
            int gm = mBase + mOff + (mi << 4) + r + hi;
            if (gm < N) {
#pragma unroll
                for (int ni = 0; ni < 4; ++ni) {
                    int gn = nBase + nOff + (ni << 4) + nc;
                    if (gn < N) So[gm * N + gn] = acc[mi][ni][r] * scale;
                }
            }
        }
}

// ============ K3: row softmax, fp32 scores -> f16 probabilities =============
__global__ __launch_bounds__(256)
void softmax_rows(const float* __restrict__ S, _Float16* __restrict__ P, int N) {
    __shared__ float red[256];
    const int tid = threadIdx.x;
    const float* s = S + (size_t)blockIdx.x * N;
    _Float16*    p = P + (size_t)blockIdx.x * N;

    float mx = -3.0e38f;
    for (int j = tid; j < N; j += 256) mx = fmaxf(mx, s[j]);
    red[tid] = mx; __syncthreads();
    for (int o = 128; o > 0; o >>= 1) { if (tid < o) red[tid] = fmaxf(red[tid], red[tid + o]); __syncthreads(); }
    mx = red[0]; __syncthreads();

    float sum = 0.f;
    for (int j = tid; j < N; j += 256) sum += __expf(s[j] - mx);
    red[tid] = sum; __syncthreads();
    for (int o = 128; o > 0; o >>= 1) { if (tid < o) red[tid] += red[tid + o]; __syncthreads(); }
    float inv = 1.0f / red[0];

    for (int j = tid; j < N; j += 256) p[j] = (_Float16)(__expf(s[j] - mx) * inv);
}

// ============ K4: y = P x V, scatter back to dense [bt,c,h,w] f16 ===========
// A tile (P, f16, contiguous rows) is staged with CDNA5 async global->LDS DMA.
__global__ __launch_bounds__(256)
void attn_pv_kernel(const _Float16* __restrict__ P, const _Float16* __restrict__ vf,
                    _Float16* __restrict__ yf, int N, int D, int chOff, int perT,
                    int outW, int winW, int winH, int aSh, int wSh) {
    __shared__ _Float16 As[2 * A_TILE];
    __shared__ _Float16 Bs[2 * B_TILE];

    const int tid  = threadIdx.x;
    const int lane = tid & 31;
    const int wave = tid >> 5;
    const int mOff = (wave >> 1) << 5;
    const int nOff = (wave & 1) << 6;

    const int bIdx  = blockIdx.z;
    const int img0  = bIdx * T4;
    const int mBase = blockIdx.y << 7;   // query rows
    const int nBase = blockIdx.x << 7;   // feature dim

    const _Float16* Pb = P + (size_t)bIdx * N * N;

    auto stage = [&](int k0, int buf) {
        _Float16* Ad = As + buf * A_TILE;
        _Float16* Bd = Bs + buf * B_TILE;
#pragma unroll
        for (int i = 0; i < 2; ++i) {             // A: P rows, async DMA 16B/lane
            int idx = tid + (i << 8);
            int row = idx >> 2, c8 = (idx & 3) << 3;
            int gr = mBase + row; if (gr >= N) gr = N - 1;
            const _Float16* g = Pb + (size_t)gr * N + k0 + c8;
            unsigned lds_off = (unsigned)(uintptr_t)(Ad + row * LDA + c8);
            unsigned long long ga = (unsigned long long)(uintptr_t)g;
            asm volatile("global_load_async_to_lds_b128 %0, %1, off"
                         :: "v"(lds_off), "v"(ga) : "memory");
        }
#pragma unroll
        for (int i = 0; i < 16; ++i) {            // B: V rows (gather)
            int idx = tid + (i << 8);
            int kk = idx >> 7, nn = idx & 127;
            Bd[kk * LDB + nn] =
                vf[win_addr(k0 + kk, nBase + nn, img0, chOff, perT, outW, winW, winH, aSh, wSh)];
        }
    };

    v8f acc[2][4] = {};
    stage(0, 0);
    asm volatile("s_wait_asynccnt 0" ::: "memory");
    __syncthreads();
    const int nT = N / 32;
    for (int t = 0; t < nT; ++t) {
        if (t + 1 < nT) stage(32 * (t + 1), (t + 1) & 1);
        mma_step(As + (t & 1) * A_TILE, Bs + (t & 1) * B_TILE, mOff, nOff, lane, acc);
        asm volatile("s_wait_asynccnt 0" ::: "memory");
        __syncthreads();
    }

    const int hi = (lane >> 4) << 3;
    const int nc = lane & 15;
#pragma unroll
    for (int mi = 0; mi < 2; ++mi)
#pragma unroll
        for (int r = 0; r < 8; ++r) {
            int gm = mBase + mOff + (mi << 4) + r + hi;
            if (gm < N) {
#pragma unroll
                for (int ni = 0; ni < 4; ++ni) {
                    int gn = nBase + nOff + (ni << 4) + nc;
                    yf[win_addr(gm, gn, img0, chOff, perT, outW, winW, winH, aSh, wSh)] =
                        (_Float16)acc[mi][ni][r];
                }
            }
        }
}

// ============ K5: 3x3 conv (implicit GEMM, K=2304) + bias + LeakyReLU =======
__global__ __launch_bounds__(256)
void conv3_kernel(const _Float16* __restrict__ yf, const float* __restrict__ wo,
                  const float* __restrict__ bo, float* __restrict__ out) {
    __shared__ _Float16 As[2 * A_TILE];
    __shared__ _Float16 Bs[2 * B_TILE];

    const int tid  = threadIdx.x;
    const int lane = tid & 31;
    const int wave = tid >> 5;
    const int mOff = (wave >> 1) << 5;
    const int nOff = (wave & 1) << 6;

    const int img   = blockIdx.z;
    const int mBase = blockIdx.y << 7;
    const int nBase = blockIdx.x << 7;
    const _Float16* Yi = yf + img * C256 * HW;

    auto stage = [&](int k0, int buf) {
        _Float16* Ad = As + buf * A_TILE;
        _Float16* Bd = Bs + buf * B_TILE;
#pragma unroll
        for (int i = 0; i < 4; ++i) {             // A: wo flat [256][2304] fp32 -> f16
            int idx = tid + (i << 8);
            int row = idx >> 3, c4 = (idx & 7) << 2;
            float4 f = *(const float4*)(wo + (mBase + row) * CONVK + k0 + c4);
            st4h(&Ad[row * LDA + c4], f);
        }
#pragma unroll
        for (int i = 0; i < 16; ++i) {            // B: shifted image rows (gather + zero pad)
            int idx = tid + (i << 8);
            int kk = idx >> 7, nn = idx & 127;
            int k  = k0 + kk;
            int ci = k / 9;
            int r9 = k - ci * 9;
            int kh = r9 / 3;
            int kw = r9 - kh * 3;
            int gn = nBase + nn;
            int h  = gn / 96;
            int w  = gn - h * 96;
            int ih = h + kh - 1, iw = w + kw - 1;
            _Float16 v = (_Float16)0.f;
            if ((unsigned)ih < 96u && (unsigned)iw < 96u)
                v = Yi[ci * HW + ih * 96 + iw];
            Bd[kk * LDB + nn] = v;
        }
    };

    v8f acc[2][4] = {};
    stage(0, 0);
    __syncthreads();
    const int nT = CONVK / 32;
    for (int t = 0; t < nT; ++t) {
        if (t + 1 < nT) {
            __builtin_prefetch(wo + mBase * CONVK + 32 * (t + 1), 0, 1);
            stage(32 * (t + 1), (t + 1) & 1);
        }
        mma_step(As + (t & 1) * A_TILE, Bs + (t & 1) * B_TILE, mOff, nOff, lane, acc);
        __syncthreads();
    }

    float* Oi = out + img * C256 * HW;
    const int hi = (lane >> 4) << 3;
    const int nc = lane & 15;
#pragma unroll
    for (int mi = 0; mi < 2; ++mi)
#pragma unroll
        for (int r = 0; r < 8; ++r) {
            int gm = mBase + mOff + (mi << 4) + r + hi;
            float bb = bo[gm];
#pragma unroll
            for (int ni = 0; ni < 4; ++ni) {
                float v = acc[mi][ni][r] + bb;
                Oi[gm * HW + nBase + nOff + (ni << 4) + nc] = (v >= 0.f) ? v : 0.2f * v;
            }
        }
}

// ============================ launch ========================================
extern "C" void kernel_launch(void* const* d_in, const int* in_sizes, int n_in,
                              void* d_out, int out_size, void* d_ws, size_t ws_size,
                              hipStream_t stream) {
    const float* x  = (const float*)d_in[0];
    const float* wq = (const float*)d_in[1];
    const float* bq = (const float*)d_in[2];
    const float* wk = (const float*)d_in[3];
    const float* bk = (const float*)d_in[4];
    const float* wv = (const float*)d_in[5];
    const float* bv = (const float*)d_in[6];
    const float* wo = (const float*)d_in[7];
    const float* bo = (const float*)d_in[8];
    float* out = (float*)d_out;

    // workspace partition
    char* ws = (char*)d_ws;
    size_t off = 0;
    auto carve = [&](size_t bytes) -> void* {
        void* p = ws + off;
        off += (bytes + 255) & ~(size_t)255;
        return p;
    };
    const size_t QKV_B = (size_t)BT * C256 * HW * sizeof(_Float16);   // 37.75 MB
    _Float16* qf = (_Float16*)carve(QKV_B);
    _Float16* kf = (_Float16*)carve(QKV_B);
    _Float16* vf = (_Float16*)carve(QKV_B);
    _Float16* yf = (_Float16*)carve(QKV_B);
    float*    s0 = (float*)   carve((size_t)2 * N0 * N0 * sizeof(float));
    _Float16* p0 = (_Float16*)carve((size_t)2 * N0 * N0 * sizeof(_Float16));
    float*    s1 = (float*)   carve((size_t)2 * N1 * N1 * sizeof(float));
    _Float16* p1 = (_Float16*)carve((size_t)2 * N1 * N1 * sizeof(_Float16));

    // 1) QKV projections (f16 out)
    qkv_gemm_kernel<<<dim3(HW / 128, C256 / 128, 3 * BT), 256, 0, stream>>>(
        x, wq, bq, wk, bk, wv, bv, qf, kf, vf);

    // 2) attention scores
    // branch 0: 4x4 windows, chOff=0,  perT=576 (24*24), outW=24, aSh=4, wSh=2
    attn_scores_kernel<<<dim3((N0 + 127) / 128, (N0 + 127) / 128, 2), 256, 0, stream>>>(
        qf, kf, s0, N0, D0, 0, 576, 24, 4, 4, 4, 2, 0.022097086912079608f);
    // branch 1: 8x8 windows, chOff=128, perT=144 (12*12), outW=12, aSh=6, wSh=3
    attn_scores_kernel<<<dim3((N1 + 127) / 128, (N1 + 127) / 128, 2), 256, 0, stream>>>(
        qf, kf, s1, N1, D1, 128, 144, 12, 8, 8, 6, 3, 0.011048543456039804f);

    // 3) softmax -> f16 probabilities
    softmax_rows<<<dim3(2 * N0), 256, 0, stream>>>(s0, p0, N0);
    softmax_rows<<<dim3(2 * N1), 256, 0, stream>>>(s1, p1, N1);

    // 4) y = P x V, scattered to dense f16 [bt,c,h,w]
    attn_pv_kernel<<<dim3(D0 / 128, (N0 + 127) / 128, 2), 256, 0, stream>>>(
        p0, vf, yf, N0, D0, 0, 576, 24, 4, 4, 4, 2);
    attn_pv_kernel<<<dim3(D1 / 128, (N1 + 127) / 128, 2), 256, 0, stream>>>(
        p1, vf, yf, N1, D1, 128, 144, 12, 8, 8, 6, 3);

    // 5) 3x3 conv + bias + LeakyReLU (fp32 out)
    conv3_kernel<<<dim3(HW / 128, C256 / 128, BT), 256, 0, stream>>>(yf, wo, bo, out);
}
